// CustomInt8Linear_1546188226972
// MI455X (gfx1250) — matrix operand order repaired and checked by reading, not compiled, not run on hardware
//
#include <hip/hip_runtime.h>
#include <hip/hip_bf16.h>
#include <stdint.h>

typedef _Float16 v16h __attribute__((ext_vector_type(16)));
typedef _Float16 v8h  __attribute__((ext_vector_type(8)));
typedef _Float16 v4h  __attribute__((ext_vector_type(4)));
typedef float    v8f  __attribute__((ext_vector_type(8)));

#define TOKENS 4096
#define KDIM   4096
#define NDIM   16384

#define TILE_M 128
#define TILE_N 128
#define TILE_K 64
// f16 elements per LDS row: 64 data + 16 pad = 80 (160B = 40 dwords -> conflict-free b128 reads)
#define LDS_STRIDE 80

__global__ __launch_bounds__(256)
void int8_linear_wmma(const float* __restrict__ x,
                      const int8_t* __restrict__ qw,
                      const float* __restrict__ scale_p,
                      const float* __restrict__ bias,
                      float* __restrict__ out)
{
    __shared__ __align__(16) _Float16 lds_a[TILE_M * LDS_STRIDE]; // 20 KB
    __shared__ __align__(16) _Float16 lds_b[TILE_N * LDS_STRIDE]; // 20 KB

    const int t    = threadIdx.x;
    const int lane = t & 31;
    const int wv   = t >> 5;        // wave id 0..7
    const int wm   = (wv & 1) * 64; // wave M offset inside C tile
    const int wn   = (wv >> 1) * 32;// wave N offset inside C tile
    const int lr   = lane & 15;
    const int lh   = lane >> 4;     // half-wave selector

    const int m0 = blockIdx.y * TILE_M;
    const int n0 = blockIdx.x * TILE_N;

    // loop-invariant staging coordinates
    const int arow = t >> 4;            // + i*16, i=0..7
    const int acol = (t & 15) << 2;     // f32 col
    const int brow = t >> 2;            // + i*64, i=0..1
    const int bcol = (t & 3) << 4;      // i8 col

    const float*  xbase = x  + (size_t)(m0 + arow) * KDIM + acol;
    const int8_t* qbase = qw + (size_t)(n0 + brow) * KDIM + bcol;

    v8f acc[4][2] = {};

    // register staging buffers (software pipeline, single LDS buffer)
    float4 aReg[8];
    int4   bReg[2];

    // ---- prologue: fetch tile k0 = 0 into registers
    #pragma unroll
    for (int i = 0; i < 8; ++i)
        aReg[i] = *(const float4*)(xbase + (size_t)i * 16 * KDIM);
    #pragma unroll
    for (int i = 0; i < 2; ++i)
        bReg[i] = *(const int4*)(qbase + (size_t)i * 64 * KDIM);

    for (int k0 = 0; k0 < KDIM; k0 += TILE_K) {
        __syncthreads();   // previous iteration's LDS reads complete

        // ---- store staged registers -> LDS (f32->f16, i8->f16 conversion)
        #pragma unroll
        for (int i = 0; i < 8; ++i) {
            float4 v = aReg[i];
            v4h h = { (_Float16)v.x, (_Float16)v.y, (_Float16)v.z, (_Float16)v.w };
            *(v4h*)(&lds_a[(arow + i * 16) * LDS_STRIDE + acol]) = h;
        }
        #pragma unroll
        for (int i = 0; i < 2; ++i) {
            union { int4 v; int8_t b[16]; } u;
            u.v = bReg[i];
            v8h lo, hi;
            #pragma unroll
            for (int j = 0; j < 8; ++j) {
                lo[j] = (_Float16)(int)u.b[j];
                hi[j] = (_Float16)(int)u.b[8 + j];
            }
            *(v8h*)(&lds_b[(brow + i * 64) * LDS_STRIDE + bcol    ]) = lo;
            *(v8h*)(&lds_b[(brow + i * 64) * LDS_STRIDE + bcol + 8]) = hi;
        }

        __syncthreads();   // staged tile visible to all waves

        // ---- issue NEXT tile's global loads; latency hides behind WMMA below
        const int kn = k0 + TILE_K;
        if (kn < KDIM) {
            #pragma unroll
            for (int i = 0; i < 8; ++i)
                aReg[i] = *(const float4*)(xbase + (size_t)i * 16 * KDIM + kn);
            #pragma unroll
            for (int i = 0; i < 2; ++i)
                bReg[i] = *(const int4*)(qbase + (size_t)i * 64 * KDIM + kn);
            // L2 prefetch two tiles ahead (global_prefetch_b8)
            if (kn + TILE_K < KDIM) {
                __builtin_prefetch(xbase + (size_t)(t & 7) * 16 * KDIM + kn + TILE_K, 0, 1);
                __builtin_prefetch(qbase + (size_t)(t & 1) * 64 * KDIM + kn + TILE_K, 0, 1);
            }
        }

        // ---- two 16x16x32 WMMA K-steps; batch all fragment loads first
        #pragma unroll
        for (int kk = 0; kk < TILE_K; kk += 32) {
            v16h bf[2], af[4];
            #pragma unroll
            for (int nt = 0; nt < 2; ++nt) {
                // lane -> column N = lr, K kk+lh*8 (.. +7) and kk+16+lh*8 (.. +7)
                const _Float16* bp = &lds_b[(wn + nt * 16 + lr) * LDS_STRIDE + kk + lh * 8];
                v8h b0 = *(const v8h*)bp;
                v8h b1 = *(const v8h*)(bp + 16);
                bf[nt] = __builtin_shufflevector(b0, b1,
                    0,1,2,3,4,5,6,7,8,9,10,11,12,13,14,15);
            }
            #pragma unroll
            for (int mt = 0; mt < 4; ++mt) {
                // lane -> row M = lr, K kk+lh*8 (.. +7) and kk+16+lh*8 (.. +7)
                const _Float16* ap = &lds_a[(wm + mt * 16 + lr) * LDS_STRIDE + kk + lh * 8];
                v8h a0 = *(const v8h*)ap;
                v8h a1 = *(const v8h*)(ap + 16);
                af[mt] = __builtin_shufflevector(a0, a1,
                    0,1,2,3,4,5,6,7,8,9,10,11,12,13,14,15);
            }
            #pragma unroll
            for (int mt = 0; mt < 4; ++mt)
                #pragma unroll
                for (int nt = 0; nt < 2; ++nt)
                    acc[mt][nt] = __builtin_amdgcn_wmma_f32_16x16x32_f16(
                        false, af[mt], false, bf[nt],
                        (short)0, acc[mt][nt], false, false);
        }
    }

    // ---- epilogue: y = acc * weight_scale + bias
    const float s = *scale_p;
    #pragma unroll
    for (int nt = 0; nt < 2; ++nt) {
        const int n = n0 + wn + nt * 16 + lr;
        const float bv = bias[n];
        #pragma unroll
        for (int mt = 0; mt < 4; ++mt) {
            #pragma unroll
            for (int g = 0; g < 8; ++g) {
                const int m = m0 + wm + mt * 16 + lh * 8 + g;
                out[(size_t)m * NDIM + n] = acc[mt][nt][g] * s + bv;
            }
        }
    }
}

extern "C" void kernel_launch(void* const* d_in, const int* in_sizes, int n_in,
                              void* d_out, int out_size, void* d_ws, size_t ws_size,
                              hipStream_t stream) {
    (void)in_sizes; (void)n_in; (void)out_size; (void)d_ws; (void)ws_size;
    const float*   x     = (const float*)d_in[0];
    const int8_t*  qw    = (const int8_t*)d_in[1];
    const float*   scale = (const float*)d_in[2];
    const float*   bias  = (const float*)d_in[3];
    float*         out   = (float*)d_out;

    dim3 grid(NDIM / TILE_N, TOKENS / TILE_M); // (128, 32)
    int8_linear_wmma<<<grid, 256, 0, stream>>>(x, qw, scale, bias, out);
}
